// dist_cal_19748259627382
// MI455X (gfx1250) — matrix operand order repaired and checked by reading, not compiled, not run on hardware
//
#include <hip/hip_runtime.h>

typedef __attribute__((ext_vector_type(16))) _Float16 v16h;
typedef __attribute__((ext_vector_type(8)))  _Float16 v8h;
typedef __attribute__((ext_vector_type(4)))  _Float16 v4h;
typedef __attribute__((ext_vector_type(8)))  float    v8f;

namespace {
constexpr int kD    = 512;    // feature dim (K)
constexpr int kB    = 16384;  // batch (M)
constexpr int kC    = 1000;   // classes (N, real)
constexpr int kCpad = 1024;   // classes padded to tile multiple
constexpr int BM = 256, BN = 128, BK = 32;
constexpr int LSTR = 40;      // LDS row stride in f16 (80B: 16B-aligned, bank-spread)
constexpr int kSteps = kD / BK;
// dynamic LDS: [2 bufs][hi/lo] A(256xLSTR) + [2 bufs][hi/lo] B(128xLSTR), f16
constexpr size_t kSmemA = (size_t)2 * 2 * BM * LSTR * sizeof(_Float16);  // 81920
constexpr size_t kSmemB = (size_t)2 * 2 * BN * LSTR * sizeof(_Float16);  // 40960
constexpr size_t kSmem  = kSmemA + kSmemB;                               // 122880
}

union V16 { v16h v; v8h h[2]; };

// 16B global -> LDS async copy (ASYNCcnt-tracked, no VGPR round trip).
// vdst VGPR = LDS byte offset (low 32 bits of the flat LDS pointer == the
// offset the DS unit adds to LDS_BASE); vaddr = 64-bit global address.
__device__ __forceinline__ void cp16_async(const _Float16* g, _Float16* l) {
  const unsigned lds_off = (unsigned)(unsigned long long)l;
  const unsigned long long gaddr = (unsigned long long)g;
  asm volatile("global_load_async_to_lds_b128 %0, %1, off"
               :
               : "v"(lds_off), "v"(gaddr)
               : "memory");
}

__device__ __forceinline__ void wait_async_lds() {
  asm volatile("s_wait_asynccnt 0x0" ::: "memory");
}

// ---------------------------------------------------------------------------
// Prologue: fp32 -> (f16 hi, f16 lo) split + per-row squared L2 norm.
// One wave32 per row (D=512 -> 4 float4 chunks per lane). Pad rows zeroed.
// ---------------------------------------------------------------------------
__global__ __launch_bounds__(256) void split_hilo_kernel(
    const float* __restrict__ src, _Float16* __restrict__ hi,
    _Float16* __restrict__ lo, float* __restrict__ sq,
    int rows_src, int rows_pad)
{
  const int wave = threadIdx.x >> 5;
  const int lane = threadIdx.x & 31;
  const int row  = blockIdx.x * 8 + wave;
  if (row >= rows_pad) return;

  v4h* h4 = (v4h*)(hi + (size_t)row * kD);
  v4h* l4 = (v4h*)(lo + (size_t)row * kD);
  float acc = 0.f;

  if (row < rows_src) {
    const float4* s4 = (const float4*)(src + (size_t)row * kD);
#pragma unroll
    for (int i = 0; i < (kD / 4) / 32; ++i) {
      const int idx = lane + i * 32;
      const float4 v = s4[idx];
      const _Float16 h0 = (_Float16)v.x, h1 = (_Float16)v.y;
      const _Float16 h2 = (_Float16)v.z, h3 = (_Float16)v.w;
      const _Float16 l0 = (_Float16)(v.x - (float)h0);
      const _Float16 l1 = (_Float16)(v.y - (float)h1);
      const _Float16 l2 = (_Float16)(v.z - (float)h2);
      const _Float16 l3 = (_Float16)(v.w - (float)h3);
      const v4h hv = {h0, h1, h2, h3};
      const v4h lv = {l0, l1, l2, l3};
      h4[idx] = hv;
      l4[idx] = lv;
      acc += v.x * v.x + v.y * v.y + v.z * v.z + v.w * v.w;
    }
  } else {
    const v4h z = {(_Float16)0, (_Float16)0, (_Float16)0, (_Float16)0};
#pragma unroll
    for (int i = 0; i < (kD / 4) / 32; ++i) {
      const int idx = lane + i * 32;
      h4[idx] = z;
      l4[idx] = z;
    }
  }
#pragma unroll
  for (int off = 16; off > 0; off >>= 1) acc += __shfl_xor(acc, off, 32);
  if (lane == 0) sq[row] = acc;
}

// ---------------------------------------------------------------------------
// Main kernel: cross = Xhi*Phi^T + Xhi*Plo^T + Xlo*Phi^T via f16 WMMA,
// fused epilogue dist = xsq + psq - 2*cross.
// Workgroup 256 thr = 8 waves; block tile 256x128; wave tile 64x64 (4x4 WMMA).
// Double-buffered LDS, async global->LDS staging, 1 barrier per K-step.
// ---------------------------------------------------------------------------
__global__ __launch_bounds__(256) void dist_wmma_kernel(
    const _Float16* __restrict__ xhi, const _Float16* __restrict__ xlo,
    const _Float16* __restrict__ phi, const _Float16* __restrict__ plo,
    const float* __restrict__ xsq, const float* __restrict__ psq,
    float* __restrict__ out)
{
  extern __shared__ char smem[];
  _Float16* sA = (_Float16*)smem;                   // [buf][h][BM*LSTR]
  _Float16* sB = sA + (size_t)2 * 2 * BM * LSTR;    // [buf][h][BN*LSTR]

  const int tid  = threadIdx.x;
  const int lane = tid & 31;
  const int wave = tid >> 5;
  const int wm = wave >> 1;  // 0..3 -> M offset wm*64
  const int wn = wave & 1;   // 0..1 -> N offset wn*64
  const int r  = lane & 15;  // row/col within 16-wide fragment
  const int kg = lane >> 4;  // K-group select (lane half)

  const int m0 = blockIdx.y * BM;
  const int n0 = blockIdx.x * BN;

  auto SA = [&](int buf, int h) { return sA + (size_t)(buf * 2 + h) * BM * LSTR; };
  auto SB = [&](int buf, int h) { return sB + (size_t)(buf * 2 + h) * BN * LSTR; };

  // Stage one K-step's 4 tiles into buffer `buf`. 16B pieces, striped over 256
  // threads: A = 256 rows x 4 pieces, B = 128 rows x 4 pieces.
  auto stage = [&](int buf, int k0) {
    const size_t aOff = (size_t)m0 * kD + k0;
    const size_t bOff = (size_t)n0 * kD + k0;
    _Float16* sAh = SA(buf, 0);
    _Float16* sAl = SA(buf, 1);
    _Float16* sBh = SB(buf, 0);
    _Float16* sBl = SB(buf, 1);
#pragma unroll
    for (int t = 0; t < 4; ++t) {
      const int p = tid + 256 * t;
      const int row = p >> 2, q = (p & 3) * 8;
      const size_t g = aOff + (size_t)row * kD + q;
      const int l = row * LSTR + q;
      cp16_async(xhi + g, sAh + l);
      cp16_async(xlo + g, sAl + l);
    }
#pragma unroll
    for (int t = 0; t < 2; ++t) {
      const int p = tid + 256 * t;
      const int row = p >> 2, q = (p & 3) * 8;
      const size_t g = bOff + (size_t)row * kD + q;
      const int l = row * LSTR + q;
      cp16_async(phi + g, sBh + l);
      cp16_async(plo + g, sBl + l);
    }
  };

  const v8f vzero = {0.f, 0.f, 0.f, 0.f, 0.f, 0.f, 0.f, 0.f};
  v8f acc[4][4];
#pragma unroll
  for (int mi = 0; mi < 4; ++mi)
#pragma unroll
    for (int ni = 0; ni < 4; ++ni) acc[mi][ni] = vzero;

  stage(0, 0);  // prefetch first K-step

  for (int ks = 0; ks < kSteps; ++ks) {
    const int buf = ks & 1;
    wait_async_lds();   // this wave's staged data has landed in LDS
    __syncthreads();    // everyone's landed; everyone done reading buf^1
    if (ks + 1 < kSteps) stage(buf ^ 1, (ks + 1) * BK);  // overlap next loads

    // A fragments (16x32 f16): lane holds row M=r; VGPR0-3 <- K [kg*16,kg*16+8),
    // VGPR4-7 <- K [kg*16+8, kg*16+16).
    V16 aF[2][4];
#pragma unroll
    for (int h = 0; h < 2; ++h) {
      const _Float16* s = SA(buf, h);
#pragma unroll
      for (int mi = 0; mi < 4; ++mi) {
        const int row = wm * 64 + mi * 16 + r;
        aF[h][mi].h[0] = *(const v8h*)&s[row * LSTR + kg * 16];
        aF[h][mi].h[1] = *(const v8h*)&s[row * LSTR + kg * 16 + 8];
      }
    }
    // B fragments (32x16 f16): lane holds col N=r; 16 contiguous K at kg*16.
    const _Float16* sbh = SB(buf, 0);
    const _Float16* sbl = SB(buf, 1);
#pragma unroll
    for (int ni = 0; ni < 4; ++ni) {
      const int col = wn * 64 + ni * 16 + r;
      V16 bh, bl;
      bh.h[0] = *(const v8h*)&sbh[col * LSTR + kg * 16];
      bh.h[1] = *(const v8h*)&sbh[col * LSTR + kg * 16 + 8];
      bl.h[0] = *(const v8h*)&sbl[col * LSTR + kg * 16];
      bl.h[1] = *(const v8h*)&sbl[col * LSTR + kg * 16 + 8];
#pragma unroll
      for (int mi = 0; mi < 4; ++mi) {
        acc[mi][ni] = __builtin_amdgcn_wmma_f32_16x16x32_f16(
            false, aF[0][mi].v, false, bh.v, (short)0, acc[mi][ni], false, false);
        acc[mi][ni] = __builtin_amdgcn_wmma_f32_16x16x32_f16(
            false, aF[0][mi].v, false, bl.v, (short)0, acc[mi][ni], false, false);
        acc[mi][ni] = __builtin_amdgcn_wmma_f32_16x16x32_f16(
            false, aF[1][mi].v, false, bh.v, (short)0, acc[mi][ni], false, false);
      }
    }
  }

  // Epilogue: C/D layout -> VGPR v holds M = v + 8*kg (lane halves), N = r.
#pragma unroll
  for (int mi = 0; mi < 4; ++mi) {
    float xs[8];
#pragma unroll
    for (int v = 0; v < 8; ++v)
      xs[v] = xsq[m0 + wm * 64 + mi * 16 + kg * 8 + v];
#pragma unroll
    for (int ni = 0; ni < 4; ++ni) {
      const int gc = n0 + wn * 64 + ni * 16 + r;
      if (gc < kC) {
        const float ps = psq[gc];
#pragma unroll
        for (int v = 0; v < 8; ++v) {
          const int gm = m0 + wm * 64 + mi * 16 + kg * 8 + v;
          out[(size_t)gm * kC + gc] = xs[v] + ps - 2.0f * acc[mi][ni][v];
        }
      }
    }
  }
}

// ---------------------------------------------------------------------------
extern "C" void kernel_launch(void* const* d_in, const int* in_sizes, int n_in,
                              void* d_out, int out_size, void* d_ws, size_t ws_size,
                              hipStream_t stream)
{
  (void)in_sizes; (void)n_in; (void)out_size; (void)ws_size;
  const float* x = (const float*)d_in[0];
  const float* p = (const float*)d_in[1];
  float* out = (float*)d_out;

  char* ws = (char*)d_ws;
  size_t off = 0;
  auto carve = [&](size_t bytes) -> void* {
    void* ptr = ws + off;
    off += (bytes + 255) & ~(size_t)255;
    return ptr;
  };
  _Float16* xhi = (_Float16*)carve((size_t)kB * kD * sizeof(_Float16));
  _Float16* xlo = (_Float16*)carve((size_t)kB * kD * sizeof(_Float16));
  _Float16* phi = (_Float16*)carve((size_t)kCpad * kD * sizeof(_Float16));
  _Float16* plo = (_Float16*)carve((size_t)kCpad * kD * sizeof(_Float16));
  float* xsq = (float*)carve((size_t)kB * sizeof(float));
  float* psq = (float*)carve((size_t)kCpad * sizeof(float));

  split_hilo_kernel<<<kB / 8, 256, 0, stream>>>(x, xhi, xlo, xsq, kB, kB);
  split_hilo_kernel<<<kCpad / 8, 256, 0, stream>>>(p, phi, plo, psq, kC, kCpad);
  dist_wmma_kernel<<<dim3(kCpad / BN, kB / BM), 256, kSmem, stream>>>(
      xhi, xlo, phi, plo, xsq, psq, out);
}
